// TransformerBlock_88751204205083
// MI455X (gfx1250) — matrix operand order repaired and checked by reading, not compiled, not run on hardware
//
#include <hip/hip_runtime.h>
#include <math.h>

// ---------------- problem constants ----------------
#define DIM      1024
#define N_HEADS  16
#define N_KV     4
#define HEAD_DIM 64
#define HIDDEN   2048
#define N_EXP    8
#define TOPK     2
#define BB       2
#define TT       1024
#define NTOK     (BB*TT)           // 2048 tokens
#define NSLOT    (NTOK*TOPK)       // 4096 expert assignments
#define SLOT_PAD 128               // tile overrun padding

// ---------------- WMMA types / helpers ----------------
typedef __bf16 bf16x16 __attribute__((ext_vector_type(16)));
typedef float  f32x8   __attribute__((ext_vector_type(8)));

__device__ __forceinline__ f32x8 wmma_bf16(bf16x16 a, bf16x16 b, f32x8 c) {
    return __builtin_amdgcn_wmma_f32_16x16x32_bf16(false, a, false, b, (short)0, c, false, false);
}

// Async global->LDS 16B copy (CDNA5 GLOBAL_LOAD_ASYNC_TO_LDS_B128, ASYNCcnt).
// Generic LDS pointers map to the LDS byte offset in their low 32 bits (ISA 10.2).
__device__ __forceinline__ void async_copy_b128(void* lds_ptr, const void* gptr) {
    unsigned off = (unsigned)(uintptr_t)lds_ptr;
    asm volatile("global_load_async_to_lds_b128 %0, %1, off"
                 :: "v"(off), "v"(gptr) : "memory");
}
__device__ __forceinline__ void async_wait0() {
    asm volatile("s_wait_asynccnt 0x0" ::: "memory");
}

// A fragment: 16x32 bf16. lanes0-15: row m, K[0..7] in v0..3, K[16..23] in v4..7;
// lanes16-31: row m, K[8..15] / K[24..31]. (ISA 7.12.2)
__device__ __forceinline__ bf16x16 load_a(const __bf16* base, int ldk, int mBase, int kBase) {
    const int lane = threadIdx.x & 31;
    const __bf16* p = base + (size_t)(mBase + (lane & 15)) * ldk + kBase + ((lane >> 4) * 8);
    union { uint4 q[2]; bf16x16 v; } u;
    u.q[0] = *(const uint4*)p;          // K = koff..koff+7
    u.q[1] = *(const uint4*)(p + 16);   // K = 16+koff..
    return u.v;
}

// B fragment: 32x16 bf16. lane(n = l&15); lanes0-15 hold K 0..7 / 16..23, lanes16-31 K 8..15 / 24..31,
// 2 K-values packed per VGPR (same K order as A).
__device__ __forceinline__ bf16x16 load_b(const __bf16* base, int ldk, int nBase, int kBase) {
    const int lane = threadIdx.x & 31;
    const __bf16* p = base + (size_t)(nBase + (lane & 15)) * ldk + kBase + ((lane >> 4) * 8);
    union { uint4 q[2]; bf16x16 v; } u;
    u.q[0] = *(const uint4*)p;
    u.q[1] = *(const uint4*)(p + 16);
    return u.v;
}

// ---------------- rmsnorm (writes bf16 and optionally f32) ----------------
__global__ __launch_bounds__(256) void rmsnorm_kernel(const float* __restrict__ x,
                                                      const float* __restrict__ w,
                                                      __bf16* __restrict__ ob,
                                                      float* __restrict__ of) {
    const int row = blockIdx.x;
    const float* xr = x + (size_t)row * DIM;
    __shared__ float red[256];
    float s = 0.f;
    for (int j = threadIdx.x; j < DIM; j += 256) { float v = xr[j]; s += v * v; }
    red[threadIdx.x] = s; __syncthreads();
    for (int off = 128; off > 0; off >>= 1) {
        if (threadIdx.x < off) red[threadIdx.x] += red[threadIdx.x + off];
        __syncthreads();
    }
    const float inv = rsqrtf(red[0] / (float)DIM + 1e-6f);
    for (int j = threadIdx.x; j < DIM; j += 256) {
        float y = w[j] * xr[j] * inv;
        ob[(size_t)row * DIM + j] = (__bf16)y;
        if (of) of[(size_t)row * DIM + j] = y;
    }
}

// ---------------- generic GEMM: C[M,N] = A(bf16)[M,K] * W(f32)[N,K]^T (+res) ----------------
#define BM 128
#define BN 64
#define BK 32

__global__ __launch_bounds__(256) void gemm_kernel(const __bf16* __restrict__ A, int lda,
                                                   const float* __restrict__ W, int ldb,
                                                   float* __restrict__ C, int ldc,
                                                   const float* __restrict__ res,
                                                   int K) {
    __shared__ alignas(16) __bf16 As[BM][BK];
    __shared__ alignas(16) __bf16 Bs[BN][BK];
    const int tid = threadIdx.x, lane = tid & 31, wid = tid >> 5;
    const int wm = wid >> 1, wn = wid & 1;
    const int m0 = blockIdx.x * BM, n0 = blockIdx.y * BN;

    f32x8 acc[2][2] = {};
    for (int k0 = 0; k0 < K; k0 += BK) {
#pragma unroll
        for (int it = 0; it < 2; ++it) {            // 128x32 bf16: async global->LDS
            int c = tid + it * 256;
            int r = c >> 2, co = (c & 3) * 8;
            async_copy_b128(&As[r][co], &A[(size_t)(m0 + r) * lda + k0 + co]);
        }
        {                                            // 64x32 f32 -> bf16 (VGPR path)
            int r = tid >> 2, co = (tid & 3) * 8;
            const float* wp = &W[(size_t)(n0 + r) * ldb + k0 + co];
            __builtin_prefetch(wp + BK, 0, 1);
#pragma unroll
            for (int j = 0; j < 8; ++j) Bs[r][co + j] = (__bf16)wp[j];
        }
        async_wait0();
        __syncthreads();
        bf16x16 a0 = load_a(&As[0][0], BK, wm * 32 + 0, 0);
        bf16x16 a1 = load_a(&As[0][0], BK, wm * 32 + 16, 0);
        bf16x16 b0 = load_b(&Bs[0][0], BK, wn * 32 + 0, 0);
        bf16x16 b1 = load_b(&Bs[0][0], BK, wn * 32 + 16, 0);
        acc[0][0] = wmma_bf16(a0, b0, acc[0][0]);
        acc[0][1] = wmma_bf16(a0, b1, acc[0][1]);
        acc[1][0] = wmma_bf16(a1, b0, acc[1][0]);
        acc[1][1] = wmma_bf16(a1, b1, acc[1][1]);
        __syncthreads();
    }
#pragma unroll
    for (int si = 0; si < 2; ++si)
#pragma unroll
        for (int sj = 0; sj < 2; ++sj) {
            int col = n0 + wn * 32 + sj * 16 + (lane & 15);
            int rbase = m0 + wm * 32 + si * 16 + ((lane >> 4) ? 8 : 0);
#pragma unroll
            for (int i = 0; i < 8; ++i) {
                int row = rbase + i;
                float v = acc[si][sj][i];
                if (res) v += res[(size_t)row * ldc + col];
                C[(size_t)row * ldc + col] = v;
            }
        }
}

// ---------------- RoPE + layout ----------------
__global__ __launch_bounds__(256) void rope_q_kernel(const float* __restrict__ qr,
                                                     const float* __restrict__ cosb,
                                                     const float* __restrict__ sinb,
                                                     __bf16* __restrict__ qo) {
    int idx = blockIdx.x * 256 + threadIdx.x;       // B*T*H*32 = 1M threads
    int i = idx & 31, h = (idx >> 5) & 15, t = (idx >> 9) & 1023, b = idx >> 19;
    float c = cosb[t * 32 + i], s = sinb[t * 32 + i];
    size_t n = (size_t)b * TT + t;
    float xe = qr[n * (N_HEADS * HEAD_DIM) + h * HEAD_DIM + 2 * i];
    float xo = qr[n * (N_HEADS * HEAD_DIM) + h * HEAD_DIM + 2 * i + 1];
    size_t o = (((size_t)(b * N_HEADS + h)) * TT + t) * HEAD_DIM + 2 * i;
    qo[o] = (__bf16)(xe * c - xo * s);
    qo[o + 1] = (__bf16)(xe * s + xo * c);
}

__global__ __launch_bounds__(256) void rope_kv_kernel(const float* __restrict__ kr,
                                                      const float* __restrict__ vr,
                                                      const float* __restrict__ cosb,
                                                      const float* __restrict__ sinb,
                                                      __bf16* __restrict__ ko,
                                                      __bf16* __restrict__ vt) {
    int idx = blockIdx.x * 256 + threadIdx.x;       // B*T*KV*32 = 256K threads
    int i = idx & 31, kv = (idx >> 5) & 3, t = (idx >> 7) & 1023, b = idx >> 17;
    float c = cosb[t * 32 + i], s = sinb[t * 32 + i];
    size_t n = (size_t)b * TT + t;
    float xe = kr[n * (N_KV * HEAD_DIM) + kv * HEAD_DIM + 2 * i];
    float xo = kr[n * (N_KV * HEAD_DIM) + kv * HEAD_DIM + 2 * i + 1];
    size_t o = (((size_t)(b * N_KV + kv)) * TT + t) * HEAD_DIM + 2 * i;
    ko[o] = (__bf16)(xe * c - xo * s);
    ko[o + 1] = (__bf16)(xe * s + xo * c);
    // V^T: [b][kv][d][t]
    float v0 = vr[n * (N_KV * HEAD_DIM) + kv * HEAD_DIM + 2 * i];
    float v1 = vr[n * (N_KV * HEAD_DIM) + kv * HEAD_DIM + 2 * i + 1];
    size_t vbase = ((size_t)(b * N_KV + kv)) * HEAD_DIM * TT;
    vt[vbase + (size_t)(2 * i) * TT + t] = (__bf16)v0;
    vt[vbase + (size_t)(2 * i + 1) * TT + t] = (__bf16)v1;
}

// ---------------- flash attention (64-query blocks, causal, GQA) ----------------
__global__ __launch_bounds__(128) void attn_kernel(const __bf16* __restrict__ Qb,
                                                   const __bf16* __restrict__ Kb,
                                                   const __bf16* __restrict__ Vt,
                                                   __bf16* __restrict__ Ao) {
    const int qblk = blockIdx.x, h = blockIdx.y, b = blockIdx.z;
    const int kvh = h >> 2;
    const int tid = threadIdx.x, lane = tid & 31, wid = tid >> 5;
    const int wm = wid >> 1, wn = wid & 1;

    const __bf16* Q = Qb + (((size_t)(b * N_HEADS + h)) * TT + qblk * 64) * HEAD_DIM;
    const __bf16* Kp = Kb + ((size_t)(b * N_KV + kvh)) * TT * HEAD_DIM;
    const __bf16* Vp = Vt + ((size_t)(b * N_KV + kvh)) * HEAD_DIM * TT;

    __shared__ alignas(16) __bf16 Qs[64][64];
    __shared__ alignas(16) __bf16 Ks[64][64];
    __shared__ alignas(16) __bf16 Vs[64][64];
    __shared__ float Ss[64][64];
    __shared__ alignas(16) __bf16 Ps[64][64];
    __shared__ float mrow[64], lrow[64], arow[64];

#pragma unroll
    for (int it = 0; it < 4; ++it) {                // 64x64 bf16: async global->LDS
        int c = tid + it * 128;
        int r = c >> 3, co = (c & 7) * 8;
        async_copy_b128(&Qs[r][co], &Q[(size_t)r * 64 + co]);
    }
    if (tid < 64) { mrow[tid] = -INFINITY; lrow[tid] = 0.f; }
    f32x8 o[2][2] = {};
    const float scale = 0.125f;                      // 1/sqrt(64)

    for (int kb = 0; kb <= qblk; ++kb) {
#pragma unroll
        for (int it = 0; it < 4; ++it) {
            int c = tid + it * 128;
            int r = c >> 3, co = (c & 7) * 8;
            async_copy_b128(&Ks[r][co], &Kp[(size_t)(kb * 64 + r) * 64 + co]);
            async_copy_b128(&Vs[r][co], &Vp[(size_t)r * TT + kb * 64 + co]);
        }
        async_wait0();
        __syncthreads();
        // S = Q K^T (64x64, K-dim = 64 -> 2 wmma steps)
        f32x8 s[2][2] = {};
#pragma unroll
        for (int ks = 0; ks < 2; ++ks) {
            bf16x16 a0 = load_a(&Qs[0][0], 64, wm * 32 + 0, ks * 32);
            bf16x16 a1 = load_a(&Qs[0][0], 64, wm * 32 + 16, ks * 32);
            bf16x16 b0 = load_b(&Ks[0][0], 64, wn * 32 + 0, ks * 32);
            bf16x16 b1 = load_b(&Ks[0][0], 64, wn * 32 + 16, ks * 32);
            s[0][0] = wmma_bf16(a0, b0, s[0][0]);
            s[0][1] = wmma_bf16(a0, b1, s[0][1]);
            s[1][0] = wmma_bf16(a1, b0, s[1][0]);
            s[1][1] = wmma_bf16(a1, b1, s[1][1]);
        }
#pragma unroll
        for (int si = 0; si < 2; ++si)
#pragma unroll
            for (int sj = 0; sj < 2; ++sj) {
                int col = wn * 32 + sj * 16 + (lane & 15);
                int rbase = wm * 32 + si * 16 + ((lane >> 4) ? 8 : 0);
#pragma unroll
                for (int i = 0; i < 8; ++i) {
                    int row = rbase + i;
                    int qg = qblk * 64 + row, kg = kb * 64 + col;
                    Ss[row][col] = (kg <= qg) ? s[si][sj][i] * scale : -INFINITY;
                }
            }
        __syncthreads();
        // online softmax per query row
        if (tid < 64) {
            int r = tid;
            float mb = -INFINITY;
            for (int c2 = 0; c2 < 64; ++c2) mb = fmaxf(mb, Ss[r][c2]);
            float mold = mrow[r];
            float mnew = fmaxf(mold, mb);
            float al = (mold == -INFINITY) ? 0.f : __expf(mold - mnew);
            float sum = 0.f;
            for (int c2 = 0; c2 < 64; ++c2) {
                float p = __expf(Ss[r][c2] - mnew);
                Ps[r][c2] = (__bf16)p;
                sum += p;
            }
            lrow[r] = lrow[r] * al + sum;
            mrow[r] = mnew; arow[r] = al;
        }
        __syncthreads();
        // rescale O, then O += P * V^T
#pragma unroll
        for (int si = 0; si < 2; ++si) {
            int rbase = wm * 32 + si * 16 + ((lane >> 4) ? 8 : 0);
#pragma unroll
            for (int sj = 0; sj < 2; ++sj)
#pragma unroll
                for (int i = 0; i < 8; ++i) o[si][sj][i] *= arow[rbase + i];
        }
#pragma unroll
        for (int ks = 0; ks < 2; ++ks) {
            bf16x16 a0 = load_a(&Ps[0][0], 64, wm * 32 + 0, ks * 32);
            bf16x16 a1 = load_a(&Ps[0][0], 64, wm * 32 + 16, ks * 32);
            bf16x16 b0 = load_b(&Vs[0][0], 64, wn * 32 + 0, ks * 32);
            bf16x16 b1 = load_b(&Vs[0][0], 64, wn * 32 + 16, ks * 32);
            o[0][0] = wmma_bf16(a0, b0, o[0][0]);
            o[0][1] = wmma_bf16(a0, b1, o[0][1]);
            o[1][0] = wmma_bf16(a1, b0, o[1][0]);
            o[1][1] = wmma_bf16(a1, b1, o[1][1]);
        }
        __syncthreads();
    }
    // epilogue: divide by l, store bf16 to [b][t][h*64+d]
#pragma unroll
    for (int si = 0; si < 2; ++si)
#pragma unroll
        for (int sj = 0; sj < 2; ++sj) {
            int col = wn * 32 + sj * 16 + (lane & 15);
            int rbase = wm * 32 + si * 16 + ((lane >> 4) ? 8 : 0);
#pragma unroll
            for (int i = 0; i < 8; ++i) {
                int row = rbase + i;
                int t = qblk * 64 + row;
                float v = o[si][sj][i] / lrow[row];
                Ao[((size_t)(b * TT) + t) * (N_HEADS * HEAD_DIM) + h * HEAD_DIM + col] = (__bf16)v;
            }
        }
}

// ---------------- MoE routing ----------------
__global__ void init_kernel(int* counts, int* etok) {
    int i = blockIdx.x * 256 + threadIdx.x;
    if (i < N_EXP) counts[i] = 0;
    if (i < NSLOT + SLOT_PAD) etok[i] = 0;
}

__global__ __launch_bounds__(256) void router_kernel(const float* __restrict__ xn2f,
                                                     const float* __restrict__ rw,
                                                     const float* __restrict__ rb,
                                                     int* __restrict__ counts,
                                                     int* __restrict__ tok_e,
                                                     int* __restrict__ tok_r,
                                                     float* __restrict__ tok_w) {
    const int lane = threadIdx.x & 31, wid = threadIdx.x >> 5;
    const int n = blockIdx.x * 8 + wid;
    const float* xr = xn2f + (size_t)n * DIM;
    float lg[N_EXP];
#pragma unroll
    for (int e = 0; e < N_EXP; ++e) {
        const float* wr = rw + (size_t)e * DIM;
        float p = 0.f;
        for (int j = lane; j < DIM; j += 32) p += xr[j] * wr[j];
        for (int m = 16; m > 0; m >>= 1) p += __shfl_xor(p, m, 32);
        lg[e] = p + rb[e];
    }
    if (lane == 0) {
        float mx = lg[0];
        for (int e = 1; e < N_EXP; ++e) mx = fmaxf(mx, lg[e]);
        float pe[N_EXP], sum = 0.f;
        for (int e = 0; e < N_EXP; ++e) { pe[e] = __expf(lg[e] - mx); sum += pe[e]; }
        for (int e = 0; e < N_EXP; ++e) pe[e] /= sum;
        int i0 = 0;
        for (int e = 1; e < N_EXP; ++e) if (pe[e] > pe[i0]) i0 = e;
        int i1 = (i0 == 0) ? 1 : 0;
        for (int e = 0; e < N_EXP; ++e) if (e != i0 && pe[e] > pe[i1]) i1 = e;
        float wsum = pe[i0] + pe[i1] + 1e-9f;
        int r0 = atomicAdd(&counts[i0], 1);
        int r1 = atomicAdd(&counts[i1], 1);
        tok_e[2 * n] = i0; tok_e[2 * n + 1] = i1;
        tok_r[2 * n] = r0; tok_r[2 * n + 1] = r1;
        tok_w[2 * n] = pe[i0] / wsum; tok_w[2 * n + 1] = pe[i1] / wsum;
    }
}

__global__ void offs_kernel(const int* counts, int* offs) {
    if (threadIdx.x == 0 && blockIdx.x == 0) {
        int acc = 0;
        for (int e = 0; e < N_EXP; ++e) { offs[e] = acc; acc += counts[e]; }
    }
}

__global__ void build_kernel(const int* __restrict__ tok_e, const int* __restrict__ tok_r,
                             const int* __restrict__ offs,
                             int* __restrict__ etok, int* __restrict__ assign) {
    int idx = blockIdx.x * 256 + threadIdx.x;      // NSLOT
    if (idx >= NSLOT) return;
    int n = idx >> 1;
    int e = tok_e[idx];
    int slot = offs[e] + tok_r[idx];
    etok[slot] = n;
    assign[idx] = slot;
}

// ---------------- MoE expert GEMMs ----------------
__global__ __launch_bounds__(256) void moe_gateup_kernel(const __bf16* __restrict__ Xn,
                                                         const float* __restrict__ Wg,
                                                         const float* __restrict__ Wu,
                                                         const int* __restrict__ counts,
                                                         const int* __restrict__ offs,
                                                         const int* __restrict__ etok,
                                                         __bf16* __restrict__ H) {
    const int e = blockIdx.z;
    const int cnt = counts[e];
    const int m0 = blockIdx.x * BM;
    if (m0 >= cnt) return;
    const int n0 = blockIdx.y * BN;
    const int slot0 = offs[e] + m0;
    const float* wg = Wg + (size_t)e * HIDDEN * DIM;
    const float* wu = Wu + (size_t)e * HIDDEN * DIM;

    __shared__ alignas(16) __bf16 As[BM][BK];
    __shared__ alignas(16) __bf16 Bg[BN][BK];
    __shared__ alignas(16) __bf16 Bu[BN][BK];
    __shared__ int toks[BM];
    const int tid = threadIdx.x, lane = tid & 31, wid = tid >> 5;
    const int wm = wid >> 1, wn = wid & 1;
    if (tid < BM) toks[tid] = etok[slot0 + tid];
    __syncthreads();

    f32x8 ga[2][2] = {}, ua[2][2] = {};
    for (int k0 = 0; k0 < DIM; k0 += BK) {
#pragma unroll
        for (int it = 0; it < 2; ++it) {            // gathered A rows, async copy
            int c = tid + it * 256;
            int r = c >> 2, co = (c & 3) * 8;
            async_copy_b128(&As[r][co], &Xn[(size_t)toks[r] * DIM + k0 + co]);
        }
        {
            int r = tid >> 2, co = (tid & 3) * 8;
            const float* p1 = &wg[(size_t)(n0 + r) * DIM + k0 + co];
            const float* p2 = &wu[(size_t)(n0 + r) * DIM + k0 + co];
#pragma unroll
            for (int j = 0; j < 8; ++j) { Bg[r][co + j] = (__bf16)p1[j]; Bu[r][co + j] = (__bf16)p2[j]; }
        }
        async_wait0();
        __syncthreads();
        bf16x16 a0 = load_a(&As[0][0], BK, wm * 32 + 0, 0);
        bf16x16 a1 = load_a(&As[0][0], BK, wm * 32 + 16, 0);
        bf16x16 g0 = load_b(&Bg[0][0], BK, wn * 32 + 0, 0);
        bf16x16 g1 = load_b(&Bg[0][0], BK, wn * 32 + 16, 0);
        bf16x16 u0 = load_b(&Bu[0][0], BK, wn * 32 + 0, 0);
        bf16x16 u1 = load_b(&Bu[0][0], BK, wn * 32 + 16, 0);
        ga[0][0] = wmma_bf16(a0, g0, ga[0][0]);
        ga[0][1] = wmma_bf16(a0, g1, ga[0][1]);
        ga[1][0] = wmma_bf16(a1, g0, ga[1][0]);
        ga[1][1] = wmma_bf16(a1, g1, ga[1][1]);
        ua[0][0] = wmma_bf16(a0, u0, ua[0][0]);
        ua[0][1] = wmma_bf16(a0, u1, ua[0][1]);
        ua[1][0] = wmma_bf16(a1, u0, ua[1][0]);
        ua[1][1] = wmma_bf16(a1, u1, ua[1][1]);
        __syncthreads();
    }
#pragma unroll
    for (int si = 0; si < 2; ++si)
#pragma unroll
        for (int sj = 0; sj < 2; ++sj) {
            int col = n0 + wn * 32 + sj * 16 + (lane & 15);
            int rbase = wm * 32 + si * 16 + ((lane >> 4) ? 8 : 0);
#pragma unroll
            for (int i = 0; i < 8; ++i) {
                int r = rbase + i;
                if (m0 + r < cnt) {
                    float g = ga[si][sj][i], u = ua[si][sj][i];
                    float hv = (g / (1.f + __expf(-g))) * u;   // silu(g)*u
                    H[(size_t)(slot0 + r) * HIDDEN + col] = (__bf16)hv;
                }
            }
        }
}

__global__ __launch_bounds__(256) void moe_down_kernel(const __bf16* __restrict__ H,
                                                       const float* __restrict__ Wd,
                                                       const int* __restrict__ counts,
                                                       const int* __restrict__ offs,
                                                       float* __restrict__ eo) {
    const int e = blockIdx.z;
    const int cnt = counts[e];
    const int m0 = blockIdx.x * BM;
    if (m0 >= cnt) return;
    const int n0 = blockIdx.y * BN;
    const int slot0 = offs[e] + m0;
    const float* wd = Wd + (size_t)e * DIM * HIDDEN;

    __shared__ alignas(16) __bf16 As[BM][BK];
    __shared__ alignas(16) __bf16 Bs[BN][BK];
    const int tid = threadIdx.x, lane = tid & 31, wid = tid >> 5;
    const int wm = wid >> 1, wn = wid & 1;

    f32x8 acc[2][2] = {};
    for (int k0 = 0; k0 < HIDDEN; k0 += BK) {
#pragma unroll
        for (int it = 0; it < 2; ++it) {
            int c = tid + it * 256;
            int r = c >> 2, co = (c & 3) * 8;
            async_copy_b128(&As[r][co], &H[(size_t)(slot0 + r) * HIDDEN + k0 + co]);
        }
        {
            int r = tid >> 2, co = (tid & 3) * 8;
            const float* wp = &wd[(size_t)(n0 + r) * HIDDEN + k0 + co];
#pragma unroll
            for (int j = 0; j < 8; ++j) Bs[r][co + j] = (__bf16)wp[j];
        }
        async_wait0();
        __syncthreads();
        bf16x16 a0 = load_a(&As[0][0], BK, wm * 32 + 0, 0);
        bf16x16 a1 = load_a(&As[0][0], BK, wm * 32 + 16, 0);
        bf16x16 b0 = load_b(&Bs[0][0], BK, wn * 32 + 0, 0);
        bf16x16 b1 = load_b(&Bs[0][0], BK, wn * 32 + 16, 0);
        acc[0][0] = wmma_bf16(a0, b0, acc[0][0]);
        acc[0][1] = wmma_bf16(a0, b1, acc[0][1]);
        acc[1][0] = wmma_bf16(a1, b0, acc[1][0]);
        acc[1][1] = wmma_bf16(a1, b1, acc[1][1]);
        __syncthreads();
    }
#pragma unroll
    for (int si = 0; si < 2; ++si)
#pragma unroll
        for (int sj = 0; sj < 2; ++sj) {
            int col = n0 + wn * 32 + sj * 16 + (lane & 15);
            int rbase = wm * 32 + si * 16 + ((lane >> 4) ? 8 : 0);
#pragma unroll
            for (int i = 0; i < 8; ++i) {
                int r = rbase + i;
                if (m0 + r < cnt)
                    eo[(size_t)(slot0 + r) * DIM + col] = acc[si][sj][i];
            }
        }
}

// ---------------- final combine (deterministic per-token sum) ----------------
__global__ __launch_bounds__(256) void combine_kernel(const float* __restrict__ x1,
                                                      const float* __restrict__ eo,
                                                      const int* __restrict__ assign,
                                                      const float* __restrict__ tw,
                                                      float* __restrict__ out) {
    const int n = blockIdx.x;
    const int a0 = assign[2 * n], a1 = assign[2 * n + 1];
    const float w0 = tw[2 * n], w1 = tw[2 * n + 1];
    for (int j = threadIdx.x; j < DIM; j += 256)
        out[(size_t)n * DIM + j] = x1[(size_t)n * DIM + j]
                                 + w0 * eo[(size_t)a0 * DIM + j]
                                 + w1 * eo[(size_t)a1 * DIM + j];
}

// ---------------- host orchestration ----------------
extern "C" void kernel_launch(void* const* d_in, const int* in_sizes, int n_in,
                              void* d_out, int out_size, void* d_ws, size_t ws_size,
                              hipStream_t stream) {
    (void)in_sizes; (void)n_in; (void)out_size; (void)ws_size;
    const float* x     = (const float*)d_in[0];
    const float* cosb  = (const float*)d_in[1];
    const float* sinb  = (const float*)d_in[2];
    const float* nw1   = (const float*)d_in[3];
    const float* wq    = (const float*)d_in[4];
    const float* wk    = (const float*)d_in[5];
    const float* w_v   = (const float*)d_in[6];
    const float* wo    = (const float*)d_in[7];
    const float* nw2   = (const float*)d_in[8];
    const float* rw    = (const float*)d_in[9];
    const float* rb    = (const float*)d_in[10];
    const float* wgate = (const float*)d_in[11];
    const float* wup   = (const float*)d_in[12];
    const float* wdown = (const float*)d_in[13];
    float* out = (float*)d_out;

    char* p = (char*)d_ws;
    auto alloc = [&](size_t bytes) -> void* {
        void* r = (void*)p;
        p += (bytes + 255) & ~(size_t)255;
        return r;
    };
    __bf16* xn    = (__bf16*)alloc((size_t)NTOK * DIM * 2);
    float*  q_raw = (float*) alloc((size_t)NTOK * N_HEADS * HEAD_DIM * 4);
    float*  k_raw = (float*) alloc((size_t)NTOK * N_KV * HEAD_DIM * 4);
    float*  v_raw = (float*) alloc((size_t)NTOK * N_KV * HEAD_DIM * 4);
    __bf16* qbuf  = (__bf16*)alloc((size_t)BB * N_HEADS * TT * HEAD_DIM * 2);
    __bf16* kbuf  = (__bf16*)alloc((size_t)BB * N_KV * TT * HEAD_DIM * 2);
    __bf16* vtb   = (__bf16*)alloc((size_t)BB * N_KV * HEAD_DIM * TT * 2);
    __bf16* aout  = (__bf16*)alloc((size_t)NTOK * DIM * 2);
    float*  x1    = (float*) alloc((size_t)NTOK * DIM * 4);
    __bf16* xn2b  = (__bf16*)alloc((size_t)NTOK * DIM * 2);
    float*  xn2f  = (float*) alloc((size_t)NTOK * DIM * 4);
    int*    counts= (int*)   alloc(N_EXP * 4);
    int*    offs  = (int*)   alloc(N_EXP * 4);
    int*    tok_e = (int*)   alloc(NSLOT * 4);
    int*    tok_r = (int*)   alloc(NSLOT * 4);
    float*  tok_w = (float*) alloc(NSLOT * 4);
    int*    etok  = (int*)   alloc((NSLOT + SLOT_PAD) * 4);
    int*    assign= (int*)   alloc(NSLOT * 4);
    __bf16* hbuf  = (__bf16*)alloc((size_t)(NSLOT + SLOT_PAD) * HIDDEN * 2);
    float*  eo    = (float*) alloc((size_t)NSLOT * DIM * 4);

    // 1) routing scratch init
    init_kernel<<<(NSLOT + SLOT_PAD + 255) / 256, 256, 0, stream>>>(counts, etok);
    // 2) attention-input rmsnorm
    rmsnorm_kernel<<<NTOK, 256, 0, stream>>>(x, nw1, xn, nullptr);
    // 3) Q/K/V projections (WMMA)
    gemm_kernel<<<dim3(NTOK / BM, (N_HEADS * HEAD_DIM) / BN), 256, 0, stream>>>(
        xn, DIM, wq, DIM, q_raw, N_HEADS * HEAD_DIM, nullptr, DIM);
    gemm_kernel<<<dim3(NTOK / BM, (N_KV * HEAD_DIM) / BN), 256, 0, stream>>>(
        xn, DIM, wk, DIM, k_raw, N_KV * HEAD_DIM, nullptr, DIM);
    gemm_kernel<<<dim3(NTOK / BM, (N_KV * HEAD_DIM) / BN), 256, 0, stream>>>(
        xn, DIM, w_v, DIM, v_raw, N_KV * HEAD_DIM, nullptr, DIM);
    // 4) RoPE + layout (Q/K per-head bf16, V transposed)
    rope_q_kernel<<<(BB * TT * N_HEADS * 32) / 256, 256, 0, stream>>>(q_raw, cosb, sinb, qbuf);
    rope_kv_kernel<<<(BB * TT * N_KV * 32) / 256, 256, 0, stream>>>(k_raw, v_raw, cosb, sinb, kbuf, vtb);
    // 5) causal flash attention (WMMA)
    attn_kernel<<<dim3(TT / 64, N_HEADS, BB), 128, 0, stream>>>(qbuf, kbuf, vtb, aout);
    // 6) output projection + residual: x1 = x + attn @ wo^T
    gemm_kernel<<<dim3(NTOK / BM, DIM / BN), 256, 0, stream>>>(
        aout, DIM, wo, DIM, x1, DIM, x, DIM);
    // 7) ffn-input rmsnorm (bf16 for GEMMs, f32 for router)
    rmsnorm_kernel<<<NTOK, 256, 0, stream>>>(x1, nw2, xn2b, xn2f);
    // 8) router top-2 + slot assignment
    router_kernel<<<NTOK / 8, 256, 0, stream>>>(xn2f, rw, rb, counts, tok_e, tok_r, tok_w);
    offs_kernel<<<1, 1, 0, stream>>>(counts, offs);
    build_kernel<<<NSLOT / 256, 256, 0, stream>>>(tok_e, tok_r, offs, etok, assign);
    // 9) expert gate+up (fused, gathered A) -> h = silu(g)*u
    moe_gateup_kernel<<<dim3(NTOK / BM, HIDDEN / BN, N_EXP), 256, 0, stream>>>(
        xn2b, wgate, wup, counts, offs, etok, hbuf);
    // 10) expert down projection -> per-slot output
    moe_down_kernel<<<dim3(NTOK / BM, DIM / BN, N_EXP), 256, 0, stream>>>(
        hbuf, wdown, counts, offs, eo);
    // 11) deterministic combine: out = x1 + w0*eo[slot0] + w1*eo[slot1]
    combine_kernel<<<NTOK, 256, 0, stream>>>(x1, eo, assign, tok_w, out);
}
